// EquivariantGNN_45440753991899
// MI455X (gfx1250) — matrix operand order repaired and checked by reading, not compiled, not run on hardware
//
#include <hip/hip_runtime.h>
#include <hip/hip_bf16.h>

typedef __attribute__((ext_vector_type(16))) _Float16 v16h;
typedef __attribute__((ext_vector_type(8)))  float    v8f;

#define NN 50000
#define NE 1600000
#define NH 8
#define HIDW 128
#define NGRAPH 64
#define NT_E (NE / 16)   // 100000 edge tiles
#define NT_N (NN / 16)   // 3125 node tiles

// ---- ordered-int encoding for float atomicMax ----
__device__ __forceinline__ int f2ord(float f) {
  int i = __float_as_int(f);
  return (i >= 0) ? i : (i ^ 0x7FFFFFFF);
}
__device__ __forceinline__ float ord2f(int k) {
  return __int_as_float((k >= 0) ? k : (k ^ 0x7FFFFFFF));
}

// ---------------- init scratch ----------------
__global__ void k_init(int* mkey, float* denom, float* agg, float* gsum, float* gcnt) {
  int i = blockIdx.x * blockDim.x + threadIdx.x;
  if (i < NN * NH) { mkey[i] = 0x807FFFFF; denom[i] = 0.0f; }  // -inf encoded
  if (i < NN * HIDW) agg[i] = 0.0f;
  if (i < NGRAPH) { gsum[i] = 0.0f; gcnt[i] = 0.0f; }
}

// ---------------- pack B-matrices (f32 -> f16, WMMA B lane layout) ----------------
// B tile (32x16): lane L owns column n; its 16 halves are K = kofs/2+j (j<8), 8+kofs/2+j (j>=8),
// kofs = (L<16 ? 0 : 16).
__global__ void k_pack_b(const float* Wk, const float* Wv, const float* Wo,
                         _Float16* Bqk, _Float16* Bv, _Float16* Bo) {
  int t = blockIdx.x * blockDim.x + threadIdx.x;
  if (t < 256) {                        // Bv: e_in(12,pad32) @ Wv -> [ct][lane][16]
    int ct = t >> 5, lane = t & 31;
    int kofs = (lane < 16) ? 0 : 16;
    int n = ct * 16 + (lane & 15);
    _Float16* o = Bv + (size_t)(ct * 32 + lane) * 16;
    for (int j = 0; j < 16; ++j) {
      int K = (j < 8) ? (kofs / 2 + j) : (8 + kofs / 2 + j);
      o[j] = (K < 12) ? (_Float16)Wv[K * HIDW + n] : (_Float16)0.0f;
    }
  } else if (t < 256 + 1024) {          // Bo: agg(128) @ Wo -> [kc][ct][lane][16]
    int u = t - 256;
    int kc = u >> 8, rem = u & 255, ct = rem >> 5, lane = rem & 31;
    int kofs = (lane < 16) ? 0 : 16;
    int n = ct * 16 + (lane & 15);
    _Float16* o = Bo + (size_t)((kc * 8 + ct) * 32 + lane) * 16;
    for (int j = 0; j < 16; ++j) {
      int K = kc * 32 + ((j < 8) ? (kofs / 2 + j) : (8 + kofs / 2 + j));
      o[j] = (_Float16)Wo[K * HIDW + n];
    }
  } else if (t < 256 + 1024 + 256) {    // Bqk: q_head(16,pad32) @ Wk_h^T -> [h][lane][16]
    int u = t - 1280;
    int h = u >> 5, lane = u & 31;
    int kofs = (lane < 16) ? 0 : 16;    // K index = d (head-dim), valid < 16
    int j = lane & 15;                  // output column = e_in feature j, valid < 12
    _Float16* o = Bqk + (size_t)(h * 32 + lane) * 16;
    for (int tt = 0; tt < 16; ++tt) {
      int d = (tt < 8) ? (kofs / 2 + tt) : (8 + kofs / 2 + tt);
      o[tt] = (d < 16 && j < 12) ? (_Float16)Wk[j * HIDW + h * 16 + d] : (_Float16)0.0f;
    }
  }
}

// ---------------- node features + q = x @ Wq ----------------
__global__ void k_node_q(const float* pos, const int* z, const float* emb,
                         const float* Wq, float* x, float* q) {
  int n = blockIdx.x * blockDim.x + threadIdx.x;
  if (n >= NN) return;
  float xv[8];
  xv[0] = pos[n * 3 + 0];
  xv[1] = pos[n * 3 + 1];
  xv[2] = pos[n * 3 + 2];
  const float* e = emb + (size_t)z[n] * 5;
  #pragma unroll
  for (int j = 0; j < 5; ++j) xv[3 + j] = e[j];
  #pragma unroll
  for (int j = 0; j < 8; ++j) x[(size_t)n * 8 + j] = xv[j];
  for (int o = 0; o < HIDW; ++o) {
    float acc = 0.0f;
    #pragma unroll
    for (int i = 0; i < 8; ++i) acc = fmaf(xv[i], Wq[i * HIDW + o], acc);
    q[(size_t)n * HIDW + o] = acc;
  }
}

// ---------------- qk[n][h][j] = sum_d Wk[j][h*16+d] * q[n][h*16+d]  (WMMA) ----------------
__global__ __launch_bounds__(256) void k_qk(const float* q, const _Float16* Bqk, float* qk) {
  int wave = blockIdx.x * (blockDim.x >> 5) + (threadIdx.x >> 5);
  if (wave >= NT_N) return;
  int lane = threadIdx.x & 31;
  int m = lane & 15;
  int nbase = wave * 16;
  int klo = (lane < 16) ? 0 : 8;        // lanes 0-15 supply d=0..7, lanes 16-31 d=8..15
  int half8 = (lane >= 16) ? 8 : 0;
  const v16h* Bv16 = (const v16h*)Bqk;
  for (int h = 0; h < NH; ++h) {
    v16h a = {};
    const float* qrow = q + (size_t)(nbase + m) * HIDW + h * 16;
    #pragma unroll
    for (int j = 0; j < 8; ++j) a[j] = (_Float16)qrow[j + klo];   // j>=8 -> K>=16 -> zero
    v16h b = Bv16[h * 32 + lane];
    v8f c = {};
    c = __builtin_amdgcn_wmma_f32_16x16x32_f16(false, a, false, b, (short)0, c, false, false);
    if (m < 12) {
      #pragma unroll
      for (int r = 0; r < 8; ++r)
        qk[(size_t)(nbase + r + half8) * 96 + h * 12 + m] = c[r];
    }
  }
}

// ---------------- edge pass 1: scores (lane-local) + segment max ----------------
__global__ void k_edge_scores(const int* src, const int* dst, const float* xf,
                              const float* edge_attr, const float* qk,
                              float* sc, int* mkey) {
  int e = blockIdx.x * blockDim.x + threadIdx.x;
  if (e >= NE) return;
  float ein[12];
  const float* xr = xf + (size_t)src[e] * 8;
  #pragma unroll
  for (int j = 0; j < 8; ++j) ein[j] = xr[j];
  const float* ar = edge_attr + (size_t)e * 4;
  #pragma unroll
  for (int j = 0; j < 4; ++j) ein[8 + j] = ar[j];
  int d = dst[e];
  const float* qkr = qk + (size_t)d * 96;
  int* mrow = mkey + d * NH;
  #pragma unroll
  for (int h = 0; h < NH; ++h) {
    float s = 0.0f;
    #pragma unroll
    for (int j = 0; j < 12; ++j) s = fmaf(ein[j], qkr[h * 12 + j], s);
    s *= 0.25f;                         // 1/sqrt(16)
    sc[(size_t)e * NH + h] = s;
    atomicMax(&mrow[h], f2ord(s));
  }
}

// ---------------- exp + segment denom ----------------
__global__ void k_edge_exp(const int* dst, const int* mkey, float* sc, float* denom) {
  int i = blockIdx.x * blockDim.x + threadIdx.x;
  if (i >= NE * NH) return;
  int e = i >> 3, h = i & 7;
  int d = dst[e];
  float mv = ord2f(mkey[d * NH + h]);
  float ex = __expf(sc[i] - mv);
  sc[i] = ex;                           // overwrite score with exp in place
  atomicAdd(&denom[d * NH + h], ex);
}

// ---------------- inverse denominators (turn hot-loop divide into multiply) ----------------
__global__ void k_invd(float* denom) {
  int i = blockIdx.x * blockDim.x + threadIdx.x;
  if (i >= NN * NH) return;
  denom[i] = 1.0f / fmaxf(denom[i], 1e-9f);
}

// Build A-matrix lane vector for an edge tile (16x32 f16, K>=12 zero).
// ISA A layout: lanes 0-15 hold K=0..7 (j<8) & 16..23 (j>=8); lanes 16-31 hold K=8..15 & 24..31.
__device__ __forceinline__ v16h build_edge_A(int lane, int ebase,
                                             const int* src, const float* xf,
                                             const float* edge_attr) {
  v16h a = {};
  int m = lane & 15;
  int e = ebase + m;
  if (lane < 16) {                      // K = 0..7 : x[src[e]]
    const float* xr = xf + (size_t)src[e] * 8;
    #pragma unroll
    for (int j = 0; j < 8; ++j) a[j] = (_Float16)xr[j];
  } else {                              // K = 8..11 : edge_attr[e]
    const float* er = edge_attr + (size_t)e * 4;
    #pragma unroll
    for (int j = 0; j < 4; ++j) a[j] = (_Float16)er[j];
  }
  return a;
}

// ---------------- edge pass 2: v = e_in@Wv (WMMA), weighted scatter into agg ----------------
__global__ __launch_bounds__(256) void k_edge_agg(
    const int* src, const int* dst, const float* xf, const float* edge_attr,
    const _Float16* Bvb, const float* sc, const float* invd, float* agg) {
  int wave = blockIdx.x * (blockDim.x >> 5) + (threadIdx.x >> 5);
  if (wave >= NT_E) return;
  int lane = threadIdx.x & 31;
  int m = lane & 15;
  int ebase = wave * 16;
  v16h a = build_edge_A(lane, ebase, src, xf, edge_attr);
  int half8 = (lane >= 16) ? 8 : 0;

  // hoist destination indices / base offsets out of the head loop
  int dn[8];
  #pragma unroll
  for (int r = 0; r < 8; ++r) dn[r] = dst[ebase + r + half8];

  const v16h* Bv16 = (const v16h*)Bvb;
  for (int h = 0; h < NH; ++h) {
    v16h b = Bv16[h * 32 + lane];
    v8f c = {};
    c = __builtin_amdgcn_wmma_f32_16x16x32_f16(false, a, false, b, (short)0, c, false, false);
    #pragma unroll
    for (int r = 0; r < 8; ++r) {
      int er = ebase + r + half8;
      int d = dn[r];
      float w = sc[(size_t)er * NH + h] * invd[d * NH + h];
      atomicAdd(&agg[(size_t)d * HIDW + h * 16 + m], w * c[r]);
    }
  }
}

// ---------------- node pass: feats = agg@Wo (WMMA), relu, y = h@lin_W, graph pool ----------------
__global__ __launch_bounds__(256) void k_node_out(
    const float* agg, const _Float16* Bo, const float* lin_W, const float* lin_b,
    const int* batch, float* gsum, float* gcnt) {
  int wave = blockIdx.x * (blockDim.x >> 5) + (threadIdx.x >> 5);
  if (wave >= NT_N) return;
  int lane = threadIdx.x & 31;
  int m = lane & 15;
  int nbase = wave * 16;
  const float* arow = agg + (size_t)(nbase + m) * HIDW;
  int klo = (lane < 16) ? 0 : 8;
  v16h a[4];
  #pragma unroll
  for (int kc = 0; kc < 4; ++kc) {
    #pragma unroll
    for (int j = 0; j < 16; ++j) {
      int K = kc * 32 + ((j < 8) ? (j + klo) : (j + 8 + klo));
      a[kc][j] = (_Float16)arow[K];
    }
  }
  const v16h* Bo16 = (const v16h*)Bo;
  float yacc[8] = {0.f, 0.f, 0.f, 0.f, 0.f, 0.f, 0.f, 0.f};
  for (int ct = 0; ct < 8; ++ct) {
    v8f c = {};
    #pragma unroll
    for (int kc = 0; kc < 4; ++kc) {
      v16h b = Bo16[(kc * 8 + ct) * 32 + lane];
      c = __builtin_amdgcn_wmma_f32_16x16x32_f16(false, a[kc], false, b, (short)0, c, false, false);
    }
    float lw = lin_W[ct * 16 + m];
    #pragma unroll
    for (int r = 0; r < 8; ++r) yacc[r] = fmaf(fmaxf(c[r], 0.0f), lw, yacc[r]);
  }
  int half8 = (lane >= 16) ? 8 : 0;
  #pragma unroll
  for (int r = 0; r < 8; ++r) {
    float s = yacc[r];
    s += __shfl_xor(s, 1);
    s += __shfl_xor(s, 2);
    s += __shfl_xor(s, 4);
    s += __shfl_xor(s, 8);
    if (m == r) {
      int node = nbase + r + half8;
      float y = s + lin_b[0];
      int g = batch[node];
      atomicAdd(&gsum[g], y);
      atomicAdd(&gcnt[g], 1.0f);
    }
  }
}

__global__ void k_finalize(const float* gsum, const float* gcnt, float* out) {
  int g = threadIdx.x;
  if (g < NGRAPH) out[g] = gsum[g] / fmaxf(gcnt[g], 1.0f);
}

extern "C" void kernel_launch(void* const* d_in, const int* in_sizes, int n_in,
                              void* d_out, int out_size, void* d_ws, size_t ws_size,
                              hipStream_t stream) {
  const float* pos        = (const float*)d_in[0];
  const int*   z          = (const int*)d_in[1];
  const int*   edge_index = (const int*)d_in[2];
  const float* edge_attr  = (const float*)d_in[3];
  const int*   batch      = (const int*)d_in[4];
  const float* emb        = (const float*)d_in[5];
  const float* Wq         = (const float*)d_in[6];
  const float* Wk         = (const float*)d_in[7];
  const float* Wv         = (const float*)d_in[8];
  const float* Wo         = (const float*)d_in[9];
  const float* lin_W      = (const float*)d_in[10];
  const float* lin_b      = (const float*)d_in[11];
  float* out = (float*)d_out;

  const int* src = edge_index;
  const int* dst = edge_index + NE;

  // carve workspace (256B aligned regions)
  char* w = (char*)d_ws;
  auto carve = [&](size_t bytes) {
    void* p = (void*)w;
    w += (bytes + 255) & ~(size_t)255;
    return p;
  };
  float*    x     = (float*)carve((size_t)NN * 8 * 4);
  float*    q     = (float*)carve((size_t)NN * HIDW * 4);
  float*    qk    = (float*)carve((size_t)NN * 96 * 4);
  int*      mkey  = (int*)carve((size_t)NN * NH * 4);
  float*    denom = (float*)carve((size_t)NN * NH * 4);
  float*    sc    = (float*)carve((size_t)NE * NH * 4);
  float*    agg   = (float*)carve((size_t)NN * HIDW * 4);
  _Float16* Bqk   = (_Float16*)carve((size_t)8 * 32 * 16 * 2);
  _Float16* Bv    = (_Float16*)carve((size_t)8 * 32 * 16 * 2);
  _Float16* Bo    = (_Float16*)carve((size_t)4 * 8 * 32 * 16 * 2);
  float*    gsum  = (float*)carve((size_t)NGRAPH * 4);
  float*    gcnt  = (float*)carve((size_t)NGRAPH * 4);

  k_init<<<(NN * HIDW + 255) / 256, 256, 0, stream>>>(mkey, denom, agg, gsum, gcnt);
  k_pack_b<<<6, 256, 0, stream>>>(Wk, Wv, Wo, Bqk, Bv, Bo);
  k_node_q<<<(NN + 255) / 256, 256, 0, stream>>>(pos, z, emb, Wq, x, q);
  k_qk<<<(NT_N + 7) / 8, 256, 0, stream>>>(q, Bqk, qk);
  k_edge_scores<<<(NE + 255) / 256, 256, 0, stream>>>(src, dst, x, edge_attr, qk, sc, mkey);
  k_edge_exp<<<(NE * NH + 255) / 256, 256, 0, stream>>>(dst, mkey, sc, denom);
  k_invd<<<(NN * NH + 255) / 256, 256, 0, stream>>>(denom);
  k_edge_agg<<<NT_E / 8, 256, 0, stream>>>(src, dst, x, edge_attr, Bv, sc, denom, agg);
  k_node_out<<<(NT_N + 7) / 8, 256, 0, stream>>>(agg, Bo, lin_W, lin_b, batch, gsum, gcnt);
  k_finalize<<<1, 64, 0, stream>>>(gsum, gcnt, out);
}
